// QuantizedLinear_22282290332380
// MI455X (gfx1250) — 8-bit matrix operand order repaired and checked by reading, compiled, not run on hardware
//
#include <hip/hip_runtime.h>
#include <stdint.h>

#define IN_F   4096
#define OUT_F  11008
#define BATCH  2048

#define LDS_STRIDE 80

typedef int v8i __attribute__((ext_vector_type(8)));
typedef int v4i_g __attribute__((vector_size(16)));

#define AS_GLOBAL __attribute__((address_space(1)))
#define AS_LDS    __attribute__((address_space(3)))

__device__ __forceinline__ void async_copy_b128(const void* gsrc, void* ldst) {
  __builtin_amdgcn_global_load_async_to_lds_b128(
      (AS_GLOBAL v4i_g*)(uintptr_t)gsrc,
      (AS_LDS v4i_g*)(uint32_t)(uintptr_t)ldst,
       0,  0);
}

__global__ __launch_bounds__(256) void pack_i8(const int* __restrict__ src,
                                               unsigned int* __restrict__ dst,
                                               int n4) {
  int i = blockIdx.x * 256 + threadIdx.x;
  if (i < n4) {
    int4 v = reinterpret_cast<const int4*>(src)[i];
    unsigned int p = (v.x & 0xff) | ((v.y & 0xff) << 8) |
                     ((v.z & 0xff) << 16) | ((unsigned int)(v.w & 0xff) << 24);
    dst[i] = p;
  }
}

__global__ __launch_bounds__(256) void ternary_gemm(
    const signed char* __restrict__ A8,
    const signed char* __restrict__ W8,
    const float* __restrict__ scale,
    const float* __restrict__ bias,
    float* __restrict__ out) {
  __shared__ char As[2][128 * LDS_STRIDE];
  __shared__ char Bs[2][128 * LDS_STRIDE];

  const int tid    = threadIdx.x;
  const int lane   = tid & 31;
  const int lane16 = lane & 15;
  const int hi8    = (lane >> 4) ? 8 : 0;
  const int hi16   = (lane >> 4) ? 16 : 0;
  const int wave   = tid >> 5;
  const int wm0    = (wave >> 2) * 64;
  const int wn0    = (wave & 3) * 32;
  const int m0     = blockIdx.y * 128;
  const int n0     = blockIdx.x * 128;

  const int row0 = tid >> 2;
  const int kc0  = (tid & 3) * 16;

  auto issue_async = [&](int buf, int kk) {
    const signed char* ab = A8 + (long)(m0 + row0) * IN_F + kk + kc0;
    const signed char* bb = W8 + (long)(n0 + row0) * IN_F + kk + kc0;
    async_copy_b128(ab,             &As[buf][row0 * LDS_STRIDE + kc0]);
    async_copy_b128(ab + 64 * IN_F, &As[buf][(row0 + 64) * LDS_STRIDE + kc0]);
    async_copy_b128(bb,             &Bs[buf][row0 * LDS_STRIDE + kc0]);
    async_copy_b128(bb + 64 * IN_F, &Bs[buf][(row0 + 64) * LDS_STRIDE + kc0]);
  };

  v8i acc[4][2];
#pragma unroll
  for (int i = 0; i < 4; ++i)
#pragma unroll
    for (int j = 0; j < 2; ++j) {
      v8i z = {0, 0, 0, 0, 0, 0, 0, 0};
      acc[i][j] = z;
    }

  issue_async(0, 0);

  const int KSTEPS = IN_F / 64;
  for (int ks = 0; ks < KSTEPS; ++ks) {
    const int buf = ks & 1;

    __builtin_amdgcn_s_wait_asynccnt(0);
    __syncthreads();

    if (ks + 1 < KSTEPS) issue_async(buf ^ 1, (ks + 1) * 64);
    if (ks + 2 < KSTEPS) {
      __builtin_prefetch(A8 + (long)(m0 + row0) * IN_F + (ks + 2) * 64 + kc0, 0, 0);
      __builtin_prefetch(W8 + (long)(n0 + row0) * IN_F + (ks + 2) * 64 + kc0, 0, 0);
    }

    v8i af[4];
#pragma unroll
    for (int sm = 0; sm < 4; ++sm) {
      const char* ar = &As[buf][(wm0 + sm * 16 + lane16) * LDS_STRIDE];
      unsigned long long d0 = *reinterpret_cast<const unsigned long long*>(ar + 0 + hi8);
      unsigned long long d1 = *reinterpret_cast<const unsigned long long*>(ar + 16 + hi8);
      unsigned long long d2 = *reinterpret_cast<const unsigned long long*>(ar + 32 + hi8);
      unsigned long long d3 = *reinterpret_cast<const unsigned long long*>(ar + 48 + hi8);
      af[sm][0] = (int)d0; af[sm][1] = (int)(d0 >> 32);
      af[sm][2] = (int)d1; af[sm][3] = (int)(d1 >> 32);
      af[sm][4] = (int)d2; af[sm][5] = (int)(d2 >> 32);
      af[sm][6] = (int)d3; af[sm][7] = (int)(d3 >> 32);
    }
    v8i bf[2];
#pragma unroll
    for (int sn = 0; sn < 2; ++sn) {
      const char* br = &Bs[buf][(wn0 + sn * 16 + lane16) * LDS_STRIDE];
      unsigned long long e0 = *reinterpret_cast<const unsigned long long*>(br + 0 + hi8);
      unsigned long long e1 = *reinterpret_cast<const unsigned long long*>(br + 16 + hi8);
      unsigned long long e2 = *reinterpret_cast<const unsigned long long*>(br + 32 + hi8);
      unsigned long long e3 = *reinterpret_cast<const unsigned long long*>(br + 48 + hi8);
      bf[sn][0] = (int)e0; bf[sn][1] = (int)(e0 >> 32);
      bf[sn][2] = (int)e1; bf[sn][3] = (int)(e1 >> 32);
      bf[sn][4] = (int)e2; bf[sn][5] = (int)(e2 >> 32);
      bf[sn][6] = (int)e3; bf[sn][7] = (int)(e3 >> 32);
    }

#pragma unroll
    for (int sm = 0; sm < 4; ++sm)
#pragma unroll
      for (int sn = 0; sn < 2; ++sn)
        acc[sm][sn] = __builtin_amdgcn_wmma_i32_16x16x64_iu8(
             true, af[sm],  true, bf[sn],
            acc[sm][sn],  false,  false);
  }

#pragma unroll
  for (int sn = 0; sn < 2; ++sn) {
    const int col = n0 + wn0 + sn * 16 + lane16;
    const float sc = scale[col];
    const float bi = bias[col];
#pragma unroll
    for (int sm = 0; sm < 4; ++sm) {
      const int rowBase = m0 + wm0 + sm * 16 + hi8;
#pragma unroll
      for (int r = 0; r < 8; ++r) {
        out[(long)(rowBase + r) * OUT_F + col] = (float)acc[sm][sn][r] * sc + bi;
      }
    }
  }
}

extern "C" void kernel_launch(void* const* d_in, const int* in_sizes, int n_in,
                              void* d_out, int out_size, void* d_ws, size_t ws_size,
                              hipStream_t stream) {
  const int*   input = (const int*)d_in[0];
  const int*   tw    = (const int*)d_in[1];
  const float* scl   = (const float*)d_in[2];
  const float* bia   = (const float*)d_in[3];
  float*       out   = (float*)d_out;

  char* ws = (char*)d_ws;
  const size_t wbytes = (size_t)OUT_F * IN_F;
  signed char* W8 = (signed char*)ws;
  signed char* A8 = (signed char*)(ws + wbytes);

  const int wn4 = (OUT_F * IN_F) / 4;
  const int an4 = (BATCH * IN_F) / 4;
  pack_i8<<<(wn4 + 255) / 256, 256, 0, stream>>>(tw, (unsigned int*)W8, wn4);
  pack_i8<<<(an4 + 255) / 256, 256, 0, stream>>>(input, (unsigned int*)A8, an4);

  dim3 grid(OUT_F / 128, BATCH / 128);
  ternary_gemm<<<grid, 256, 0, stream>>>(A8, W8, scl, bia, out);
}
